// AdditiveAttention_584115552769
// MI455X (gfx1250) — matrix operand order repaired and checked by reading, not compiled, not run on hardware
//
#include <hip/hip_runtime.h>
#include <cstdint>

typedef __attribute__((ext_vector_type(16))) __bf16 v16bf;
typedef __attribute__((ext_vector_type(8)))  float  v8f;

#define BATCH  32
#define SEQ    4096
#define HD     256
#define ED     512          // 2H
#define CH     128          // seq rows per block
#define NCHUNK (SEQ / CH)   // 32
#define AS_STRIDE 520       // 512 bf16 + 8 pad -> row stride 1040B, rotates banks
#define NKT    (ED / 32)    // 16 K-tiles
#define NHT    (HD / 16)    // 16 h-tiles

__device__ __forceinline__ unsigned short f2bf(float f) {
  unsigned u = __float_as_uint(f);
  u = (u + 0x7FFFu + ((u >> 16) & 1u)) >> 16;   // round-to-nearest-even
  return (unsigned short)u;
}
__device__ __forceinline__ unsigned pack2(float lo, float hi) {
  return (unsigned)f2bf(lo) | ((unsigned)f2bf(hi) << 16);
}
// branch-free tanh: 1 - 2/(e^(2x)+1); correct at +/-inf, no exec-mask branches
__device__ __forceinline__ float fast_tanh(float x) {
  float t = __expf(2.f * x);
  return 1.f - 2.f * __builtin_amdgcn_rcpf(t + 1.f);
}

// ---------------- Kernel 0: pre-pack Ua_w (f32) into per-lane bf16 B-fragment layout ----
// Bpack[kt][ht][lane][16] : lane holds N = lane&15, K = kt*32 + 16*(lane>>4) + (0..15)
__global__ void __launch_bounds__(256) packB_kernel(const float* __restrict__ Ua_w,
                                                    unsigned short* __restrict__ Bpack) {
  const int idx  = blockIdx.x * 256 + threadIdx.x;   // 0 .. NKT*NHT*32-1 (8192)
  const int lane = idx & 31;
  const int ht   = (idx >> 5) & (NHT - 1);
  const int kt   = idx >> 9;
  const int l15  = lane & 15, hi = lane >> 4;
  const float* wr = Ua_w + (size_t)(ht * 16 + l15) * ED + kt * 32 + hi * 16;
  unsigned short* dst = Bpack + (size_t)idx * 16;    // 32B per lane, coalesced
  #pragma unroll
  for (int j = 0; j < 4; ++j) {
    float4 f = *(const float4*)(wr + j * 4);
    dst[j * 4 + 0] = f2bf(f.x);
    dst[j * 4 + 1] = f2bf(f.y);
    dst[j * 4 + 2] = f2bf(f.z);
    dst[j * 4 + 3] = f2bf(f.w);
  }
}

// ---------------- Kernel 1: Qp[b][h] = dec[b]·Wa_w[h] + Wa_b[h] + Ua_b[h] ----------------
__global__ void __launch_bounds__(HD) qproj_kernel(const float* __restrict__ dec,
                                                   const float* __restrict__ Wa_w,
                                                   const float* __restrict__ Wa_b,
                                                   const float* __restrict__ Ua_b,
                                                   float* __restrict__ Qp) {
  const int b = blockIdx.x, h = threadIdx.x;
  const float* x = dec + (size_t)b * HD;
  const float* w = Wa_w + (size_t)h * HD;
  float s = Wa_b[h] + Ua_b[h];
  #pragma unroll 4
  for (int e = 0; e < HD; e += 4) {
    float4 xv = *(const float4*)(x + e);
    float4 wv = *(const float4*)(w + e);
    s += xv.x * wv.x + xv.y * wv.y + xv.z * wv.z + xv.w * wv.w;
  }
  Qp[b * HD + h] = s;
}

// ---------------- Kernel 2: per (chunk, batch) flash-style pass ----------------
__global__ void __launch_bounds__(256) attn_chunk_kernel(const float* __restrict__ enc,
                                                         const int*   __restrict__ mask,
                                                         const float* __restrict__ vvec,
                                                         const unsigned short* __restrict__ Bpack,
                                                         const float* __restrict__ Qp,
                                                         float* __restrict__ part) {
  __shared__ __align__(16) unsigned short As[CH * AS_STRIDE];  // chunk of enc, bf16 (~130KB)
  __shared__ float energy[CH];
  __shared__ float qsh[HD], vsh[HD];
  __shared__ float red[8];

  const int tid   = threadIdx.x;
  const int chunk = blockIdx.x, b = blockIdx.y;
  const int s0    = chunk * CH;
  const float* encB = enc + ((size_t)b * SEQ + s0) * ED;

  qsh[tid] = Qp[b * HD + tid];
  vsh[tid] = vvec[tid];
  if (tid < CH) energy[tid] = 0.f;

  // Stage enc chunk (128 x 512 f32) -> bf16 LDS. 64 float4 loads per thread.
  #pragma unroll 4
  for (int i = 0; i < (CH * ED / 4) / 256; ++i) {
    int idx = tid + i * 256;
    int row = idx >> 7;            // 128 float4 per row
    int c4  = (idx & 127) << 2;    // float column
    float4 f = *(const float4*)(encB + (size_t)row * ED + c4);
    *(uint2*)&As[row * AS_STRIDE + c4] = make_uint2(pack2(f.x, f.y), pack2(f.z, f.w));
  }
  __syncthreads();

  const int lane = tid & 31;
  const int wv   = tid >> 5;       // wave 0..7 -> owns h in [wv*32, wv*32+32)
  const int hb   = wv * 32;
  const int l15  = lane & 15;
  const int hi   = lane >> 4;      // 0 = lanes 0-15, 1 = lanes 16-31

  v8f acc[8][2];
  #pragma unroll
  for (int st = 0; st < 8; ++st)
    #pragma unroll
    for (int nt = 0; nt < 2; ++nt)
      #pragma unroll
      for (int r = 0; r < 8; ++r) acc[st][nt][r] = 0.f;

  union AF { uint4 q[2]; v16bf v; };

  for (int kt = 0; kt < NKT; ++kt) {
    // B fragments: pre-packed per-lane layout, 2 coalesced uint4 loads each.
    const unsigned short* bpw =
        Bpack + ((size_t)((kt * NHT + (wv << 1)) * 32 + lane)) * 16;
    AF bfrag[2];
    bfrag[0].q[0] = *(const uint4*)(bpw);
    bfrag[0].q[1] = *(const uint4*)(bpw + 8);
    bfrag[1].q[0] = *(const uint4*)(bpw + 32 * 16);
    bfrag[1].q[1] = *(const uint4*)(bpw + 32 * 16 + 8);
    // A fragments from LDS: lane holds M=l15, K runs [ka,ka+16).
    const int ka = kt * 32 + (hi << 4);
    #pragma unroll
    for (int st = 0; st < 8; ++st) {
      const unsigned short* ar = &As[(st * 16 + l15) * AS_STRIDE];
      AF a;
      a.q[0] = *(const uint4*)(ar + ka);
      a.q[1] = *(const uint4*)(ar + ka + 8);
      acc[st][0] = __builtin_amdgcn_wmma_f32_16x16x32_bf16(false, a.v, false, bfrag[0].v,
                                                           (short)0, acc[st][0], false, false);
      acc[st][1] = __builtin_amdgcn_wmma_f32_16x16x32_bf16(false, a.v, false, bfrag[1].v,
                                                           (short)0, acc[st][1], false, false);
    }
  }

  // Epilogue: energy[s] += sum_h v[h]*tanh(Qp[h] + V[s][h]).
  // D layout: element r at lane -> row = r + 8*hi, col(h) = l15.
  #pragma unroll
  for (int nt = 0; nt < 2; ++nt) {
    const int h  = hb + nt * 16 + l15;
    const float qp = qsh[h];
    const float vh = vsh[h];
    #pragma unroll
    for (int st = 0; st < 8; ++st) {
      #pragma unroll
      for (int r = 0; r < 8; ++r) {
        float t = vh * fast_tanh(acc[st][nt][r] + qp);
        t += __shfl_xor(t, 1);
        t += __shfl_xor(t, 2);
        t += __shfl_xor(t, 4);
        t += __shfl_xor(t, 8);          // reduced over the 16 h-columns in this half
        if (l15 == 0) atomicAdd(&energy[st * 16 + (hi << 3) + r], t);
      }
    }
  }
  __syncthreads();

  // Mask + chunk-local softmax statistics.
  float es = 0.f;
  if (tid < CH) {
    es = (mask[(size_t)b * SEQ + s0 + tid] != 0) ? energy[tid] : -1e10f;
    energy[tid] = es;
  }
  __syncthreads();
  if (tid < CH) {
    float m = es;
    m = fmaxf(m, __shfl_xor(m, 1));
    m = fmaxf(m, __shfl_xor(m, 2));
    m = fmaxf(m, __shfl_xor(m, 4));
    m = fmaxf(m, __shfl_xor(m, 8));
    m = fmaxf(m, __shfl_xor(m, 16));
    if (lane == 0) red[wv] = m;         // waves 0..3
  }
  __syncthreads();
  if (tid == 0) red[4] = fmaxf(fmaxf(red[0], red[1]), fmaxf(red[2], red[3]));
  __syncthreads();
  const float cm = red[4];
  if (tid < CH) {
    float p = __expf(es - cm);
    energy[tid] = p;
    float l = p;
    l += __shfl_xor(l, 1);
    l += __shfl_xor(l, 2);
    l += __shfl_xor(l, 4);
    l += __shfl_xor(l, 8);
    l += __shfl_xor(l, 16);
    if (lane == 0) red[wv] = l;
  }
  __syncthreads();
  if (tid == 0) red[5] = red[0] + red[1] + red[2] + red[3];
  __syncthreads();
  const float lsum = red[5];

  // Partial weighted sum over this chunk (enc re-read hits L2: just staged it).
  float a0 = 0.f, a1 = 0.f;
  for (int s = 0; s < CH; ++s) {
    const float ps = energy[s];
    const float* er = encB + (size_t)s * ED;
    a0 += ps * er[tid];
    a1 += ps * er[tid + 256];
  }
  float* pc = part + (size_t)(b * NCHUNK + chunk) * (ED + 2);
  pc[tid]       = a0;
  pc[tid + 256] = a1;
  if (tid == 0) { pc[ED] = cm; pc[ED + 1] = lsum; }
}

// ---------------- Kernel 3: combine chunk partials per batch ----------------
__global__ void __launch_bounds__(256) combine_kernel(const float* __restrict__ part,
                                                      float* __restrict__ out) {
  __shared__ float sm[NCHUNK], sl[NCHUNK], mg[2];
  const int b = blockIdx.x, tid = threadIdx.x;
  if (tid < NCHUNK) {
    const float* pc = part + (size_t)(b * NCHUNK + tid) * (ED + 2);
    sm[tid] = pc[ED];
    sl[tid] = pc[ED + 1];
  }
  __syncthreads();
  if (tid == 0) {
    float m = -3.4e38f;
    for (int c = 0; c < NCHUNK; ++c) m = fmaxf(m, sm[c]);
    float l = 0.f;
    for (int c = 0; c < NCHUNK; ++c) l += sl[c] * __expf(sm[c] - m);
    mg[0] = m;
    mg[1] = (l > 0.f) ? 1.f / l : 0.f;
  }
  __syncthreads();
  const float M = mg[0], invL = mg[1];
  float a0 = 0.f, a1 = 0.f;
  for (int c = 0; c < NCHUNK; ++c) {
    const float w  = __expf(sm[c] - M);
    const float* pc = part + (size_t)(b * NCHUNK + c) * (ED + 2);
    a0 += w * pc[tid];
    a1 += w * pc[tid + 256];
  }
  out[(size_t)b * ED + tid]       = a0 * invL;
  out[(size_t)b * ED + 256 + tid] = a1 * invL;
}

// ---------------- Host launcher ----------------
extern "C" void kernel_launch(void* const* d_in, const int* in_sizes, int n_in,
                              void* d_out, int out_size, void* d_ws, size_t ws_size,
                              hipStream_t stream) {
  (void)in_sizes; (void)n_in; (void)out_size; (void)ws_size;
  const float* enc  = (const float*)d_in[0];  // [B,S,2H]
  const float* dec  = (const float*)d_in[1];  // [B,1,H]
  const int*   mask = (const int*)d_in[2];    // [B,S]
  const float* vvec = (const float*)d_in[3];  // [1,H]
  const float* Ua_w = (const float*)d_in[4];  // [H,2H]
  const float* Ua_b = (const float*)d_in[5];  // [H]
  const float* Wa_w = (const float*)d_in[6];  // [H,H]
  const float* Wa_b = (const float*)d_in[7];  // [H]
  float* out = (float*)d_out;

  float*          Qp    = (float*)d_ws;                       // B*H floats (32KB)
  unsigned short* Bpack = (unsigned short*)(Qp + BATCH * HD); // NKT*NHT*32*16 bf16 (256KB)
  float*          part  = (float*)(Bpack + (size_t)NKT * NHT * 32 * 16); // B*NCHUNK*(ED+2)

  packB_kernel<<<(NKT * NHT * 32) / 256, 256, 0, stream>>>(Ua_w, Bpack);
  qproj_kernel<<<BATCH, HD, 0, stream>>>(dec, Wa_w, Wa_b, Ua_b, Qp);
  attn_chunk_kernel<<<dim3(NCHUNK, BATCH), 256, 0, stream>>>(enc, mask, vvec, Bpack, Qp, part);
  combine_kernel<<<BATCH, 256, 0, stream>>>(part, out);
}
